// SparseAttention_85409719648704
// MI455X (gfx1250) — matrix operand order repaired and checked by reading, not compiled, not run on hardware
//
#include <hip/hip_runtime.h>
#include <hip/hip_bf16.h>
#include <math.h>

// ---------------------------------------------------------------------------
// Problem constants (from reference)
// ---------------------------------------------------------------------------
static constexpr int Nn   = 1024;   // sequence length
static constexpr int DIMM = 1024;   // model dim
static constexpr int H    = 16;     // query heads
static constexpr int KH   = 4;      // kv heads
static constexpr int DH   = 64;     // head dim
static constexpr int QH   = 4;      // query heads per kv head
static constexpr int CBS  = 16;     // compression block size
static constexpr int NCB  = 64;     // number of compression blocks (N/16)
static constexpr int SBS  = 32;     // selection block size
static constexpr int NSEL = 8;      // selected blocks
static constexpr float SCALE = 0.125f;  // DH^-0.5

typedef __attribute__((ext_vector_type(16))) __bf16 v16bf;
typedef __attribute__((ext_vector_type(8)))  float  v8f;
typedef __attribute__((ext_vector_type(4)))  float  v4f;

#define WMMA_BF16(A_, B_, C_) \
  __builtin_amdgcn_wmma_f32_16x16x32_bf16(false, (A_), false, (B_), (short)0, (C_), false, false)

static __device__ inline v8f vzero8() {
  v8f z;
#pragma unroll
  for (int i = 0; i < 8; i++) z[i] = 0.f;
  return z;
}

// ---------------------------------------------------------------------------
// fp32-in / fp32-out GEMM via bf16 WMMA.  C[M,N] = act(A[M,K]@B[K,N]+bias)
// act: 0=none, 1=relu, 2=sigmoid
// REQUIRES: M % 128 == 0 (grid covers exactly), K % 64 == 0.
// Tile 128x128, BK=64, 256 threads = 8 waves; wave computes 32x64 (2x4 wmma
// per 32-K chunk, 16 wmma per iteration).  A and B tiles stored so fragment
// reads are contiguous (ds_load_b128): As[m][k], Bt[n][k] (B transposed).
// Staging uses float4 global loads, no per-element divergence.
// ---------------------------------------------------------------------------
__global__ __launch_bounds__(256) void gemm_bf16_wmma(
    const float* __restrict__ A, const float* __restrict__ B,
    const float* __restrict__ bias, float* __restrict__ C,
    int M, int Ncols, int K, int act)
{
  constexpr int BM = 128, BN = 128, BK = 64;
  __shared__ __bf16 As[BM][BK + 8];
  __shared__ __bf16 Bt[BN][BK + 8];   // transposed B tile: Bt[n][k]

  const int tid  = threadIdx.x;
  const int lane = tid & 31;
  const int wave = tid >> 5;
  const int bm = blockIdx.y * BM;
  const int bn = blockIdx.x * BN;
  const int wm = (wave >> 1) * 32;
  const int wn = (wave & 1) * 64;
  const bool fullN = (bn + BN <= Ncols);

  v8f acc[2][4];
#pragma unroll
  for (int mi = 0; mi < 2; mi++)
#pragma unroll
    for (int ni = 0; ni < 4; ni++) acc[mi][ni] = vzero8();

  const int mrow = lane & 15;
  const int kb   = (lane >> 4) * 8;
  const int kb2  = (lane >> 4) * 16;
  const int ncol = lane & 15;

  for (int k0 = 0; k0 < K; k0 += BK) {
    // --- stage A tile (128x64), float4 loads, no bounds checks ---
    for (int e = tid; e < BM * BK / 4; e += 256) {
      int r  = e >> 4;          // 16 float4 per row
      int c4 = (e & 15) << 2;
      v4f x = *(const v4f*)(A + (size_t)(bm + r) * K + k0 + c4);
      As[r][c4 + 0] = (__bf16)x[0];
      As[r][c4 + 1] = (__bf16)x[1];
      As[r][c4 + 2] = (__bf16)x[2];
      As[r][c4 + 3] = (__bf16)x[3];
    }
    // --- stage B tile transposed ---
    if (fullN) {
      for (int e = tid; e < BK * BN / 4; e += 256) {
        int r  = e >> 5;        // k row, 32 float4 per row
        int c4 = (e & 31) << 2; // n
        v4f x = *(const v4f*)(B + (size_t)(k0 + r) * Ncols + bn + c4);
        Bt[c4 + 0][r] = (__bf16)x[0];
        Bt[c4 + 1][r] = (__bf16)x[1];
        Bt[c4 + 2][r] = (__bf16)x[2];
        Bt[c4 + 3][r] = (__bf16)x[3];
      }
    } else {
      // small-N path: clamped address + branchless select (no exec divergence)
      for (int e = tid; e < BK * BN; e += 256) {
        int r = e >> 7, c = e & 127;
        int cc  = bn + c;
        int ccl = cc < Ncols ? cc : (Ncols - 1);
        float x = B[(size_t)(k0 + r) * Ncols + ccl];
        Bt[c][r] = (__bf16)(cc < Ncols ? x : 0.f);
      }
    }
    __syncthreads();

#pragma unroll
    for (int kc = 0; kc < 2; kc++) {
      v16bf aF[2], bF[4];
#pragma unroll
      for (int mi = 0; mi < 2; mi++) {
        int r = wm + mi * 16 + mrow;
#pragma unroll
        for (int i = 0; i < 8; i++) {
          aF[mi][i]     = As[r][kc * 32 + kb2 + i];
          aF[mi][8 + i] = As[r][kc * 32 + kb2 + 8 + i];
        }
      }
#pragma unroll
      for (int ni = 0; ni < 4; ni++) {
        int cn = wn + ni * 16 + ncol;
#pragma unroll
        for (int i = 0; i < 16; i++) bF[ni][i] = Bt[cn][kc * 32 + kb2 + i];
      }
#pragma unroll
      for (int mi = 0; mi < 2; mi++)
#pragma unroll
        for (int ni = 0; ni < 4; ni++)
          acc[mi][ni] = WMMA_BF16(aF[mi], bF[ni], acc[mi][ni]);
    }
    __syncthreads();
  }

  // epilogue (rows always in range by construction; cols checked only if needed)
  const int mb = (lane >> 4) * 8;
#pragma unroll
  for (int mi = 0; mi < 2; mi++)
#pragma unroll
    for (int ni = 0; ni < 4; ni++) {
      int col = bn + wn + ni * 16 + ncol;
      if (fullN || col < Ncols) {
        float badd = bias ? bias[col] : 0.f;
#pragma unroll
        for (int r = 0; r < 8; r++) {
          int row = bm + wm + mi * 16 + mb + r;
          float x = acc[mi][ni][r] + badd;
          if (act == 1)      x = fmaxf(x, 0.f);
          else if (act == 2) x = 1.f / (1.f + expf(-x));
          C[(size_t)row * Ncols + col] = x;
        }
      }
    }
}

// ---------------------------------------------------------------------------
// RoPE on q/k (split from qkv) + copy v.  q:(H,N,DH) k,v:(KH,N,DH)
// ---------------------------------------------------------------------------
__global__ void rope_split_kernel(const float* __restrict__ qkv,
                                  float* __restrict__ q,
                                  float* __restrict__ k,
                                  float* __restrict__ v)
{
  int idx = blockIdx.x * blockDim.x + threadIdx.x;
  const int ROPE_TOTAL = Nn * (H + KH) * (DH / 2);
  if (idx < ROPE_TOTAL) {
    int tok = idx / ((H + KH) * 32);
    int rem = idx % ((H + KH) * 32);
    int hd = rem >> 5, t = rem & 31;
    float inv = powf(10000.f, -2.f * (float)t / (float)DH);
    float ang = (float)tok * inv;
    float c = cosf(ang), s = sinf(ang);
    if (hd < H) {
      const float* src = qkv + (size_t)tok * 1536 + hd * DH;
      float x1 = src[2 * t], x2 = src[2 * t + 1];
      float* dst = q + ((size_t)hd * Nn + tok) * DH;
      dst[2 * t]     = x1 * c - x2 * s;
      dst[2 * t + 1] = x2 * c + x1 * s;
    } else {
      int khd = hd - H;
      const float* src = qkv + (size_t)tok * 1536 + H * DH + khd * DH;
      float x1 = src[2 * t], x2 = src[2 * t + 1];
      float* dst = k + ((size_t)khd * Nn + tok) * DH;
      dst[2 * t]     = x1 * c - x2 * s;
      dst[2 * t + 1] = x2 * c + x1 * s;
    }
  } else {
    int e = idx - ROPE_TOTAL;
    if (e < KH * Nn * DH) {
      int khd = e / (Nn * DH);
      int r = e % (Nn * DH);
      int tok = r / DH, d = r % DH;
      v[((size_t)khd * Nn + tok) * DH + d] =
          qkv[(size_t)tok * 1536 + (H + KH) * DH + khd * DH + d];
    }
  }
}

// ---------------------------------------------------------------------------
// kw/vw assembly: row = kh*NCB+cb (256 rows), col = i*DH+d (1024 cols)
// ---------------------------------------------------------------------------
__global__ void kwvw_kernel(const float* __restrict__ k, const float* __restrict__ v,
                            const float* __restrict__ k_pos, const float* __restrict__ v_pos,
                            float* __restrict__ kw, float* __restrict__ vw)
{
  int idx = blockIdx.x * blockDim.x + threadIdx.x;
  const int HALF = KH * NCB * CBS * DH;  // 262144
  if (idx >= 2 * HALF) return;
  int which = idx / HALF;
  int e = idx % HALF;
  int row = e / (CBS * DH);
  int col = e % (CBS * DH);
  int kh = row / NCB, cb = row % NCB;
  int i = col / DH, d = col % DH;
  int tok = cb * CBS + i;
  if (which == 0)
    kw[(size_t)row * (CBS * DH) + col] =
        k[((size_t)kh * Nn + tok) * DH + d] + k_pos[((size_t)kh * CBS + i) * DH + d];
  else
    vw[(size_t)row * (CBS * DH) + col] =
        v[((size_t)kh * Nn + tok) * DH + d] + v_pos[((size_t)kh * CBS + i) * DH + d];
}

// ---------------------------------------------------------------------------
// Prepend mem_kv to ck/cv -> ckf/cvf (KH,65,DH)
// ---------------------------------------------------------------------------
__global__ void build_full_kernel(const float* __restrict__ ck, const float* __restrict__ cv,
                                  const float* __restrict__ mem_kv,
                                  float* __restrict__ ckf, float* __restrict__ cvf)
{
  int idx = blockIdx.x * blockDim.x + threadIdx.x;
  const int ONE = KH * 65 * DH;
  if (idx >= 2 * ONE) return;
  int which = idx / ONE;
  int e = idx % ONE;
  int kh = e / (65 * DH);
  int r = (e / DH) % 65;
  int d = e % DH;
  float val;
  if (r == 0) val = mem_kv[(size_t)which * KH * DH + kh * DH + d];
  else        val = (which ? cv : ck)[((size_t)kh * NCB + (r - 1)) * DH + d];
  (which ? cvf : ckf)[((size_t)kh * 65 + r) * DH + d] = val;
}

// ---------------------------------------------------------------------------
// Compressed attention + importance top-k.
// Grid: (N/16, KH); block: 128 (4 waves, one per query-head in the group).
// ---------------------------------------------------------------------------
__global__ __launch_bounds__(128) void cattn_kernel(
    const float* __restrict__ q, const float* __restrict__ ckf,
    const float* __restrict__ cvf, float* __restrict__ c_out,
    int* __restrict__ idx_out)
{
  __shared__ float  S[QH][16][80];        // pre-softmax scores (j padded 65->80)
  __shared__ __bf16 P[QH][16][96];        // softmax probs (padded -> 96 for K-chunks)
  __shared__ __bf16 CVt[DH][96 + 8];      // cvf transposed: CVt[d][j], j padded to 96

  const int kh = blockIdx.y;
  const int t0 = blockIdx.x * 16;
  const int tid = threadIdx.x;
  const int lane = tid & 31;
  const int qh = tid >> 5;
  const int h = kh * QH + qh;

  const int m   = lane & 15;
  const int kb  = (lane >> 4) * 8;
  const int kb2 = (lane >> 4) * 16;
  const int nc  = lane & 15;
  const int mb  = (lane >> 4) * 8;

  // --- stage CVt (reads coalesced over d), zero the j-padding ---
  for (int e = tid; e < 65 * DH; e += 128) {
    int j = e >> 6, d = e & 63;
    CVt[d][j] = (__bf16)cvf[((size_t)kh * 65 + j) * DH + d];
  }
  for (int e = tid; e < 31 * DH; e += 128) {
    int j = 65 + (e >> 6), d = e & 63;
    CVt[d][j] = (__bf16)0.f;
  }

  // --- Q fragments: 16 tokens x 64, two K=32 chunks ---
  v16bf aF[2];
  const float* qrow = q + ((size_t)h * Nn + (t0 + m)) * DH;
#pragma unroll
  for (int c = 0; c < 2; c++)
#pragma unroll
    for (int i = 0; i < 8; i++) {
      aF[c][i]     = (__bf16)qrow[c * 32 + kb2 + i];
      aF[c][8 + i] = (__bf16)qrow[c * 32 + kb2 + 8 + i];
    }

  // --- S = (Q @ CKf^T) * SCALE, 5 j-tiles of 16; clamped loads, no divergence ---
  for (int jt = 0; jt < 5; jt++) {
    v8f acc = vzero8();
    const int j   = jt * 16 + nc;
    const int jcl = j < 65 ? j : 64;
    const float jmask = (j < 65) ? 1.f : 0.f;
#pragma unroll
    for (int c = 0; c < 2; c++) {
      v16bf bF;
#pragma unroll
      for (int i = 0; i < 16; i++) {
        float x = ckf[((size_t)kh * 65 + jcl) * DH + c * 32 + kb2 + i];
        bF[i] = (__bf16)(x * jmask);
      }
      acc = WMMA_BF16(aF[c], bF, acc);
    }
#pragma unroll
    for (int r = 0; r < 8; r++) S[qh][mb + r][jt * 16 + nc] = acc[r] * SCALE;
  }
  __syncthreads();

  // --- softmax (64 rows total: 4 qh x 16 tokens) ---
  if (tid < 64) {
    int qq = tid >> 4, mm = tid & 15;
    float mx = -1e30f;
    for (int j = 0; j < 65; j++) mx = fmaxf(mx, S[qq][mm][j]);
    float sum = 0.f;
    for (int j = 0; j < 65; j++) sum += expf(S[qq][mm][j] - mx);
    float inv = 1.f / sum;
    for (int j = 0; j < 96; j++)
      P[qq][mm][j] = (__bf16)((j < 65) ? expf(S[qq][mm][j] - mx) * inv : 0.f);
  }
  // --- importance + top-8 (one thread per token; deterministic, stable ties) ---
  if (tid < 16) {
    int mm = tid;
    float imp[32];
    for (int f = 0; f < 32; f++) {
      float s = 0.f;
      for (int qq = 0; qq < QH; qq++) s += S[qq][mm][1 + 2 * f] + S[qq][mm][2 + 2 * f];
      imp[f] = s;   // monotonic in the reference's mean; selection identical
    }
    int* op = idx_out + ((size_t)kh * Nn + (t0 + mm)) * NSEL;
    for (int s = 0; s < NSEL; s++) {
      int best = 0; float bv = -1e30f;
      for (int f = 0; f < 32; f++) if (imp[f] > bv) { bv = imp[f]; best = f; }
      op[s] = best;
      imp[best] = -1e30f;
    }
  }
  __syncthreads();

  // --- c_out = P @ CVf : 3 K-chunks (96), 4 n-tiles (64); LDS reads contiguous ---
  for (int nt = 0; nt < 4; nt++) {
    v8f acc = vzero8();
#pragma unroll
    for (int c = 0; c < 3; c++) {
      v16bf aP, bF;
#pragma unroll
      for (int i = 0; i < 8; i++) {
        aP[i]     = P[qh][m][c * 32 + kb2 + i];
        aP[8 + i] = P[qh][m][c * 32 + kb2 + 8 + i];
      }
#pragma unroll
      for (int i = 0; i < 16; i++) bF[i] = CVt[nt * 16 + nc][c * 32 + kb2 + i];
      acc = WMMA_BF16(aP, bF, acc);
    }
#pragma unroll
    for (int r = 0; r < 8; r++) {
      int tok = t0 + mb + r;
      c_out[(size_t)tok * DIMM + h * DH + nt * 16 + nc] = acc[r];
    }
  }
}

// ---------------------------------------------------------------------------
// Fine (selected) attention + gated merge.  Grid: (N, KH); block: 32 (1 wave).
// The 4 group-queries occupy rows 0..3 of a zero-padded 16-row A fragment.
// Each of the 8 selected 32-key blocks is exactly one K=32 chunk of P@V.
// ---------------------------------------------------------------------------
__global__ __launch_bounds__(32) void fattn_kernel(
    const float* __restrict__ q, const float* __restrict__ k,
    const float* __restrict__ v, const int* __restrict__ idx,
    const float* __restrict__ gates, const float* __restrict__ c_out,
    float* __restrict__ combined)
{
  __shared__ float  S[QH][NSEL * SBS];        // 4 x 256
  __shared__ __bf16 P[16][NSEL * SBS];        // padded rows 4..15 = 0
  __shared__ __bf16 Vt[DH][NSEL * SBS + 8];   // gathered V transposed: Vt[d][key]
  __shared__ int sidx[NSEL];

  const int kh = blockIdx.y;
  const int tok = blockIdx.x;
  const int lane = threadIdx.x;
  if (lane < NSEL) sidx[lane] = idx[((size_t)kh * Nn + tok) * NSEL + lane];
  __syncthreads();

  const int m   = lane & 15;
  const int kb  = (lane >> 4) * 8;
  const int kb2 = (lane >> 4) * 16;
  const int nc  = lane & 15;
  const int mb  = (lane >> 4) * 8;

  // stage gathered V transposed (float4 reads coalesced over d)
  for (int e = lane; e < NSEL * SBS * DH / 4; e += 32) {
    int kl = e >> 4;            // 0..255 selected-key index
    int d4 = (e & 15) << 2;
    int key = sidx[kl >> 5] * SBS + (kl & 31);
    v4f x = *(const v4f*)(v + ((size_t)kh * Nn + key) * DH + d4);
    Vt[d4 + 0][kl] = (__bf16)x[0];
    Vt[d4 + 1][kl] = (__bf16)x[1];
    Vt[d4 + 2][kl] = (__bf16)x[2];
    Vt[d4 + 3][kl] = (__bf16)x[3];
  }

  // Q fragment (rows 0..3 valid, padded to 16); clamped loads, branchless mask
  const int   hq    = kh * QH + (m < QH ? m : 0);
  const float qmask = (m < QH) ? 1.f : 0.f;
  const float* qr = q + ((size_t)hq * Nn + tok) * DH;
  v16bf aF[2];
#pragma unroll
  for (int c = 0; c < 2; c++)
#pragma unroll
    for (int i = 0; i < 8; i++) {
      aF[c][i]     = (__bf16)(qr[c * 32 + kb2 + i] * qmask);
      aF[c][8 + i] = (__bf16)(qr[c * 32 + kb2 + 8 + i] * qmask);
    }

  // S = Q @ Ksel^T * SCALE  (8 selected blocks x 2 n-tiles of 16 keys)
  // K rows are contiguous in global over the fragment index -> vector loads.
  for (int s = 0; s < NSEL; s++) {
    int kbase = sidx[s] * SBS;
    for (int nt = 0; nt < 2; nt++) {
      v8f acc = vzero8();
      int key = kbase + nt * 16 + nc;
#pragma unroll
      for (int c = 0; c < 2; c++) {
        v16bf bF;
#pragma unroll
        for (int i = 0; i < 16; i++)
          bF[i] = (__bf16)k[((size_t)kh * Nn + key) * DH + c * 32 + kb2 + i];
        acc = WMMA_BF16(aF[c], bF, acc);
      }
#pragma unroll
      for (int r = 0; r < 8; r++)
        if (mb + r < QH) S[mb + r][s * SBS + nt * 16 + nc] = acc[r] * SCALE;
    }
  }
  __syncthreads();

  // zero pad rows 4..15 of P
  for (int e = lane; e < 12 * 256; e += 32) P[4 + (e >> 8)][e & 255] = (__bf16)0.f;
  // softmax rows 0..3
  if (lane < QH) {
    float mx = -1e30f;
    for (int j = 0; j < 256; j++) mx = fmaxf(mx, S[lane][j]);
    float sum = 0.f;
    for (int j = 0; j < 256; j++) sum += expf(S[lane][j] - mx);
    float inv = 1.f / sum;
    for (int j = 0; j < 256; j++) P[lane][j] = (__bf16)(expf(S[lane][j] - mx) * inv);
  }
  __syncthreads();

  // f_out = P(16x256) @ Vsel(256x64); all LDS fragment reads contiguous
  for (int nt = 0; nt < 4; nt++) {
    v8f acc = vzero8();
    for (int c = 0; c < NSEL; c++) {
      v16bf aP, bF;
#pragma unroll
      for (int i = 0; i < 8; i++) {
        aP[i]     = P[m][c * 32 + kb2 + i];
        aP[8 + i] = P[m][c * 32 + kb2 + 8 + i];
      }
#pragma unroll
      for (int i = 0; i < 16; i++) bF[i] = Vt[nt * 16 + nc][c * 32 + kb2 + i];
      acc = WMMA_BF16(aP, bF, acc);
    }
#pragma unroll
    for (int r = 0; r < 8; r++) {
      int qh = mb + r;
      if (qh < QH) {
        int h = kh * QH + qh;
        int d = nt * 16 + nc;
        float g0 = gates[(size_t)tok * (2 * H) + h * 2 + 0];
        float g1 = gates[(size_t)tok * (2 * H) + h * 2 + 1];
        float co = c_out[(size_t)tok * DIMM + h * DH + d];
        combined[(size_t)tok * DIMM + h * DH + d] = g0 * co + g1 * acc[r];
      }
    }
  }
}

// ---------------------------------------------------------------------------
// Launch pipeline
// ---------------------------------------------------------------------------
extern "C" void kernel_launch(void* const* d_in, const int* in_sizes, int n_in,
                              void* d_out, int out_size, void* d_ws, size_t ws_size,
                              hipStream_t stream)
{
  (void)in_sizes; (void)n_in; (void)out_size; (void)ws_size;
  const float* inp    = (const float*)d_in[0];
  const float* w_qkv  = (const float*)d_in[1];
  const float* k_pos  = (const float*)d_in[2];
  const float* v_pos  = (const float*)d_in[3];
  const float* mem_kv = (const float*)d_in[4];
  const float* kc_w1  = (const float*)d_in[5];
  const float* kc_b1  = (const float*)d_in[6];
  const float* kc_w2  = (const float*)d_in[7];
  const float* kc_b2  = (const float*)d_in[8];
  const float* vc_w1  = (const float*)d_in[9];
  const float* vc_b1  = (const float*)d_in[10];
  const float* vc_w2  = (const float*)d_in[11];
  const float* vc_b2  = (const float*)d_in[12];
  const float* gate_w = (const float*)d_in[13];
  const float* gate_b = (const float*)d_in[14];
  const float* out_w  = (const float*)d_in[15];
  float* out = (float*)d_out;

  float* ws = (float*)d_ws;
  size_t off = 0;
  auto alloc = [&](size_t n) { float* p = ws + off; off += n; return p; };
  float* qkv   = alloc((size_t)Nn * 1536);
  float* gates = alloc((size_t)Nn * 2 * H);
  float* qb    = alloc((size_t)H * Nn * DH);
  float* kbuf  = alloc((size_t)KH * Nn * DH);
  float* vbuf  = alloc((size_t)KH * Nn * DH);
  float* kw    = alloc((size_t)KH * NCB * CBS * DH);
  float* vw    = alloc((size_t)KH * NCB * CBS * DH);
  float* khid  = alloc((size_t)KH * NCB * CBS * DH);
  float* vhid  = alloc((size_t)KH * NCB * CBS * DH);
  float* ck    = alloc((size_t)KH * NCB * DH);
  float* cv    = alloc((size_t)KH * NCB * DH);
  float* ckf   = alloc((size_t)KH * 65 * DH);
  float* cvf   = alloc((size_t)KH * 65 * DH);
  float* c_out = alloc((size_t)Nn * DIMM);
  float* comb  = alloc((size_t)Nn * DIMM);
  int*   idxb  = (int*)(ws + off);

  dim3 blk(256);

  // 1) qkv = inp @ w_qkv            (1024 x 1536, K=1024)
  gemm_bf16_wmma<<<dim3(1536 / 128, 1024 / 128), blk, 0, stream>>>(
      inp, w_qkv, nullptr, qkv, Nn, 1536, DIMM, 0);
  // 2) gates = sigmoid(inp @ gate_w + gate_b)   (1024 x 32)
  gemm_bf16_wmma<<<dim3(1, 1024 / 128), blk, 0, stream>>>(
      inp, gate_w, gate_b, gates, Nn, 2 * H, DIMM, 2);
  // 3) RoPE + split
  {
    int total = Nn * (H + KH) * (DH / 2) + KH * Nn * DH;
    rope_split_kernel<<<(total + 255) / 256, blk, 0, stream>>>(qkv, qb, kbuf, vbuf);
  }
  // 4) kw/vw assembly
  kwvw_kernel<<<(2 * KH * NCB * CBS * DH + 255) / 256, blk, 0, stream>>>(
      kbuf, vbuf, k_pos, v_pos, kw, vw);
  // 5-8) compression MLPs (256 x 1024 x 1024 -> relu -> 256 x 64)
  gemm_bf16_wmma<<<dim3(8, 2), blk, 0, stream>>>(kw, kc_w1, kc_b1, khid, 256, 1024, 1024, 1);
  gemm_bf16_wmma<<<dim3(1, 2), blk, 0, stream>>>(khid, kc_w2, kc_b2, ck, 256, 64, 1024, 0);
  gemm_bf16_wmma<<<dim3(8, 2), blk, 0, stream>>>(vw, vc_w1, vc_b1, vhid, 256, 1024, 1024, 1);
  gemm_bf16_wmma<<<dim3(1, 2), blk, 0, stream>>>(vhid, vc_w2, vc_b2, cv, 256, 64, 1024, 0);
  // 9) prepend mem slots
  build_full_kernel<<<(2 * KH * 65 * DH + 255) / 256, blk, 0, stream>>>(
      ck, cv, mem_kv, ckf, cvf);
  // 10) compressed attention + top-k selection
  cattn_kernel<<<dim3(Nn / 16, KH), dim3(128), 0, stream>>>(qb, ckf, cvf, c_out, idxb);
  // 11) fine attention + gated merge
  fattn_kernel<<<dim3(Nn, KH), dim3(32), 0, stream>>>(qb, kbuf, vbuf, idxb, gates, c_out, comb);
  // 12) output projection
  gemm_bf16_wmma<<<dim3(1024 / 128, 1024 / 128), blk, 0, stream>>>(
      comb, out_w, nullptr, out, Nn, DIMM, DIMM, 0);
}
